// AutoregressiveMatrixChain_32899449487394
// MI455X (gfx1250) — matrix operand order repaired and checked by reading, not compiled, not run on hardware
//
#include <hip/hip_runtime.h>
#include <hip/hip_bf16.h>
#include <math.h>

// ---------------------------------------------------------------------------
// CDNA5 (gfx1250, wave32) implementation of AutoregressiveMatrixChain.
// All matmuls routed through v_wmma_f32_16x16x32_bf16 (fp32 -> bf16 staged
// via LDS, f32 accumulation). Block tile 32(M) x 64(N), K-step 32, 4 waves,
// each wave computes 16x32 with two accumulators sharing one A fragment.
// Shape contract (holds for every call below): N % 64 == 0, K % 32 == 0.
// Only M is ragged (1 / 8 / 9 / 72): branchless clamp-row + zero-select on
// load, guard on store. Staging pointers are precomputed per-thread and
// bumped each K-step; loads are issued in a batch before conversion so the
// scheduler can keep all 6 global_load_b128 in flight.
// ---------------------------------------------------------------------------

typedef __attribute__((ext_vector_type(16))) __bf16 v16bf;
typedef __attribute__((ext_vector_type(8)))  float  v8f;
typedef __attribute__((ext_vector_type(4)))  float  v4f;
typedef __attribute__((ext_vector_type(4)))  __bf16 v4bf;

#define Hd   768
#define Bn   8
#define LP   512
#define LL   256
#define CN   4096
#define NS   9
#define NT   4

// d_out layout (floats): summary(8*4*768) | stopl(32) | stopp(32) | oplogits(8*4*4096) | chain(8)
#define SUM_OFF   0
#define STOPL_OFF 24576
#define STOPP_OFF 24608
#define OPL_OFF   24640
#define CL_OFF    155712

#define AS_LD 40   // bf16 elements per As row (32 + pad), 80B rows (8B aligned)
#define BS_LD 72   // bf16 elements per Bs row (64 + pad), 144B rows (8B aligned)

// ------------------------- generic batched WMMA GEMM -----------------------
// C[bz] = epilogue( scale * A[bz] (MxK) * B[bz] (KxN) + rowBias[m] + colBias[n] )
// BT=true  : B stored transposed (N x K row-major, ldb = K-stride)
// EPI==1   : tanh epilogue
template <bool BT, int EPI>
__global__ __launch_bounds__(128) void wmma_gemm(
    const float* __restrict__ A, const float* __restrict__ Bm, float* __restrict__ C,
    int M, int N, int K, int lda, int ldb, int ldc,
    long sA, long sB, long sC,
    float scale, const float* __restrict__ rowBias, const float* __restrict__ colBias)
{
    const int bz = blockIdx.z;
    const float* Ab = A  + (long)bz * sA;
    const float* Bb = Bm + (long)bz * sB;
    float*       Cb = C  + (long)bz * sC;

    const int n0 = blockIdx.x * 64;
    const int m0 = blockIdx.y * 32;
    const int tid  = threadIdx.x;
    const int wave = tid >> 5;
    const int lane = tid & 31;
    const int wm = wave >> 1;      // wave M-tile (0..1) -> rows wm*16..
    const int wn = wave & 1;       // wave N-tile (0..1) -> cols wn*32..

    __shared__ __bf16 As[32 * AS_LD];   // [m][k]
    __shared__ __bf16 Bs[32 * BS_LD];   // [k][n]

    // --- precompute per-thread staging sources/destinations (K-loop invariant)
    const float* aPtr[2]; bool aZ[2]; int aOff[2];
    #pragma unroll
    for (int i = 0; i < 2; ++i) {
        int f = tid + i * 128;                 // 256 float4s cover 32x32 A tile
        int r = f >> 3, c = (f & 7) << 2;
        int gm = m0 + r;
        int cm = (gm < M) ? gm : (M - 1);
        aPtr[i] = Ab + (long)cm * lda + c;
        aZ[i]   = (gm >= M);
        aOff[i] = r * AS_LD + c;
    }
    const float* bPtr[4]; int bOff[4]; int bN[4];
    #pragma unroll
    for (int i = 0; i < 4; ++i) {
        int f = tid + i * 128;                 // 512 float4s cover 32x64 B tile
        if (!BT) {
            int k = f >> 4, c = (f & 15) << 2;          // contiguous along n
            bPtr[i] = Bb + (long)k * ldb + n0 + c;
            bOff[i] = k * BS_LD + c;
            bN[i]   = 0;
        } else {
            int n = f >> 3, c = (f & 7) << 2;           // contiguous along k
            bPtr[i] = Bb + (long)(n0 + n) * ldb + c;
            bOff[i] = c * BS_LD;                        // row of first k element
            bN[i]   = n;
        }
    }
    const long bStep = BT ? 32 : (long)32 * ldb;

    v8f acc0 = {};
    v8f acc1 = {};

    for (int kk = 0; kk < K; kk += 32) {
        // ---- issue all staging loads first (keeps 6 b128 loads in flight)
        v4f ra[2], rb[4];
        #pragma unroll
        for (int i = 0; i < 2; ++i) ra[i] = *(const v4f*)aPtr[i];
        #pragma unroll
        for (int i = 0; i < 4; ++i) rb[i] = *(const v4f*)bPtr[i];
        #pragma unroll
        for (int i = 0; i < 2; ++i) aPtr[i] += 32;
        #pragma unroll
        for (int i = 0; i < 4; ++i) bPtr[i] += bStep;

        // ---- convert + store to LDS
        #pragma unroll
        for (int i = 0; i < 2; ++i) {
            v4f v = ra[i];
            if (aZ[i]) v = (v4f){0.f, 0.f, 0.f, 0.f};
            v4bf h;
            h.x = (__bf16)v.x; h.y = (__bf16)v.y; h.z = (__bf16)v.z; h.w = (__bf16)v.w;
            *(v4bf*)&As[aOff[i]] = h;
        }
        #pragma unroll
        for (int i = 0; i < 4; ++i) {
            v4f v = rb[i];
            if (!BT) {
                v4bf h;
                h.x = (__bf16)v.x; h.y = (__bf16)v.y; h.z = (__bf16)v.z; h.w = (__bf16)v.w;
                *(v4bf*)&Bs[bOff[i]] = h;
            } else {                                    // transpose into LDS
                Bs[bOff[i] + 0 * BS_LD + bN[i]] = (__bf16)v.x;
                Bs[bOff[i] + 1 * BS_LD + bN[i]] = (__bf16)v.y;
                Bs[bOff[i] + 2 * BS_LD + bN[i]] = (__bf16)v.z;
                Bs[bOff[i] + 3 * BS_LD + bN[i]] = (__bf16)v.w;
            }
        }
        __syncthreads();

        // ---- A fragment: lane -> M = wm*16 + lane%16 ; VGPR v covers K (ISA 7.12.2)
        v16bf af;
        const int arow = (wm * 16 + (lane & 15)) * AS_LD;
        const int grp8 = (lane >> 4) * 8;
        #pragma unroll
        for (int v = 0; v < 8; ++v) {
            int kb = (v & 4) * 4 + (v & 3) * 2 + grp8;
            af[2 * v]     = As[arow + kb];
            af[2 * v + 1] = As[arow + kb + 1];
        }
        // ---- B fragments: lane -> N = lane%16 ; element e -> K (same order as A)
        v16bf b0, b1;
        const int bbase = wn * 32 + (lane & 15);
        #pragma unroll
        for (int e = 0; e < 16; ++e) {
            int kr = ((e & 7) + ((e & 8) << 1) + grp8) * BS_LD;
            b0[e] = Bs[kr + bbase];
            b1[e] = Bs[kr + bbase + 16];
        }
        acc0 = __builtin_amdgcn_wmma_f32_16x16x32_bf16(false, af, false, b0, (short)0, acc0, false, false);
        acc1 = __builtin_amdgcn_wmma_f32_16x16x32_bf16(false, af, false, b1, (short)0, acc1, false, false);
        __syncthreads();
    }

    // ---- store: C/D layout: lane -> N = lane%16 ; VGPR r -> M = r + 8*(lane/16)
    const int grp  = lane >> 4;
    const int nn0  = n0 + wn * 32 + (lane & 15);
    #pragma unroll
    for (int r = 0; r < 8; ++r) {
        int mm = m0 + wm * 16 + grp * 8 + r;
        if (mm < M) {
            float v0 = acc0[r] * scale;
            float v1 = acc1[r] * scale;
            if (rowBias) { v0 += rowBias[mm]; v1 += rowBias[mm]; }
            if (colBias) { v0 += colBias[nn0]; v1 += colBias[nn0 + 16]; }
            if (EPI == 1) { v0 = tanhf(v0); v1 = tanhf(v1); }
            Cb[(long)mm * ldc + nn0]      = v0;
            Cb[(long)mm * ldc + nn0 + 16] = v1;
        }
    }
}

template <bool BT, int EPI>
static void gemm(hipStream_t st, const float* A, const float* Bm, float* C,
                 int M, int N, int K, int lda, int ldb, int ldc,
                 long sA, long sB, long sC, int batch,
                 float scale = 1.0f, const float* rb = nullptr, const float* cb = nullptr)
{
    dim3 g((N + 63) / 64, (M + 31) / 32, batch), b(128);
    wmma_gemm<BT, EPI><<<g, b, 0, st>>>(A, Bm, C, M, N, K, lda, ldb, ldc, sA, sB, sC, scale, rb, cb);
}

// ------------------------------ helper kernels ------------------------------

__global__ void mean_kernel(const float* __restrict__ x, float* __restrict__ out, int L) {
    int idx = blockIdx.x * blockDim.x + threadIdx.x;
    if (idx >= Bn * Hd) return;
    int b = idx / Hd, h = idx % Hd;
    float s = 0.0f;
    for (int l = 0; l < L; ++l) s += x[((long)b * L + l) * Hd + h];
    out[idx] = s / (float)L;
}

__global__ void concat_kernel(const float* __restrict__ ps, const float* __restrict__ ls,
                              float* __restrict__ cc) {
    int idx = blockIdx.x * blockDim.x + threadIdx.x;
    if (idx >= Bn * 2 * Hd) return;
    int b = idx / (2 * Hd), c = idx % (2 * Hd);
    cc[idx] = (c < Hd) ? ps[b * Hd + c] : ls[b * Hd + (c - Hd)];
}

__global__ void softmax_kernel(float* __restrict__ s, int L) {
    __shared__ float red[128];
    float* row = s + (long)blockIdx.x * L;
    float m = -1e30f;
    for (int i = threadIdx.x; i < L; i += 128) m = fmaxf(m, row[i]);
    red[threadIdx.x] = m; __syncthreads();
    for (int k = 64; k > 0; k >>= 1) {
        if (threadIdx.x < k) red[threadIdx.x] = fmaxf(red[threadIdx.x], red[threadIdx.x + k]);
        __syncthreads();
    }
    m = red[0]; __syncthreads();
    float sum = 0.0f;
    for (int i = threadIdx.x; i < L; i += 128) { float e = expf(row[i] - m); row[i] = e; sum += e; }
    red[threadIdx.x] = sum; __syncthreads();
    for (int k = 64; k > 0; k >>= 1) {
        if (threadIdx.x < k) red[threadIdx.x] += red[threadIdx.x + k];
        __syncthreads();
    }
    float inv = 1.0f / red[0];
    for (int i = threadIdx.x; i < L; i += 128) row[i] *= inv;
}

__global__ void neg_sumsq_kernel(const float* __restrict__ x, float* __restrict__ out, int K) {
    __shared__ float red[256];
    const float* row = x + (long)blockIdx.x * K;
    float s = 0.0f;
    for (int i = threadIdx.x; i < K; i += 256) { float v = row[i]; s += v * v; }
    red[threadIdx.x] = s; __syncthreads();
    for (int k = 128; k > 0; k >>= 1) {
        if (threadIdx.x < k) red[threadIdx.x] += red[threadIdx.x + k];
        __syncthreads();
    }
    if (threadIdx.x == 0) out[blockIdx.x] = -red[0];
}

__global__ void argmax_kernel(const float* __restrict__ logits, int* __restrict__ idx,
                              int Cn, int ld) {
    __shared__ float sv[256]; __shared__ int si[256];
    const float* row = logits + (long)blockIdx.x * ld;
    float best = -1e30f; int bi = 0;
    for (int i = threadIdx.x; i < Cn; i += 256) {
        float v = row[i];
        if (v > best) { best = v; bi = i; }
    }
    sv[threadIdx.x] = best; si[threadIdx.x] = bi; __syncthreads();
    for (int s = 128; s > 0; s >>= 1) {
        if (threadIdx.x < s) {
            if (sv[threadIdx.x + s] > sv[threadIdx.x] ||
                (sv[threadIdx.x + s] == sv[threadIdx.x] && si[threadIdx.x + s] < si[threadIdx.x])) {
                sv[threadIdx.x] = sv[threadIdx.x + s]; si[threadIdx.x] = si[threadIdx.x + s];
            }
        }
        __syncthreads();
    }
    if (threadIdx.x == 0) idx[blockIdx.x] = si[0];
}

__global__ void slot_seed_kernel(const float* __restrict__ ctx, const float* __restrict__ sq,
                                 float* __restrict__ seed) {
    int idx = blockIdx.x * blockDim.x + threadIdx.x;
    if (idx >= Bn * NS * Hd) return;
    int h = idx % Hd, s = (idx / Hd) % NS, b = idx / (NS * Hd);
    seed[idx] = ctx[b * Hd + h] + sq[s * Hd + h];
}

__global__ void slot_gate_kernel(const float* __restrict__ seed, const float* __restrict__ wg,
                                 const float* __restrict__ bg, float* __restrict__ probs) {
    __shared__ float red[256];
    const float* row = seed + (long)blockIdx.x * Hd;
    float s = 0.0f;
    for (int i = threadIdx.x; i < Hd; i += 256) s += row[i] * wg[i];
    red[threadIdx.x] = s; __syncthreads();
    for (int k = 128; k > 0; k >>= 1) {
        if (threadIdx.x < k) red[threadIdx.x] += red[threadIdx.x + k];
        __syncthreads();
    }
    if (threadIdx.x == 0) probs[blockIdx.x] = 1.0f / (1.0f + expf(-(red[0] + bg[0])));
}

__global__ void slot_mask_kernel(const float* __restrict__ probs, float* __restrict__ mask,
                                 float* __restrict__ cnt) {
    if (threadIdx.x != 0) return;
    bool any = false;
    for (int i = 0; i < Bn * NS; ++i) any = any || (probs[i] >= 0.5f);
    for (int b = 0; b < Bn; ++b) {
        float c = 0.0f;
        if (any) {
            for (int s = 0; s < NS; ++s) {
                float m = (probs[b * NS + s] >= 0.5f) ? 1.0f : 0.0f;
                mask[b * NS + s] = m; c += m;
            }
        } else {
            int best = 0; float bp = probs[b * NS];
            for (int s = 1; s < NS; ++s)
                if (probs[b * NS + s] > bp) { bp = probs[b * NS + s]; best = s; }
            for (int s = 0; s < NS; ++s) mask[b * NS + s] = (s == best) ? 1.0f : 0.0f;
            c = 1.0f;
        }
        cnt[b] = fmaxf(c, 1.0f);
    }
}

__global__ void matrix_summary_kernel(const float* __restrict__ cbk, const int* __restrict__ opidx,
                                      const float* __restrict__ tens, const float* __restrict__ mask,
                                      const float* __restrict__ cnt, float* __restrict__ ms,
                                      float* __restrict__ out, int t) {
    int idx = blockIdx.x * blockDim.x + threadIdx.x;
    if (idx >= Bn * Hd) return;
    int b = idx / Hd, h = idx % Hd;
    float s = 0.0f;
    for (int sl = 0; sl < NS; ++sl) s += tens[((long)b * NS + sl) * Hd + h] * mask[b * NS + sl];
    s /= cnt[b];
    float v = tanhf(cbk[(long)opidx[b] * Hd + h] + s);
    ms[idx] = v;
    out[SUM_OFF + ((long)b * NT + t) * Hd + h] = v;
}

__global__ void stop_kernel(const float* __restrict__ ctx, const float* __restrict__ ms,
                            const float* __restrict__ ws, const float* __restrict__ bs,
                            float* __restrict__ out, int t) {
    __shared__ float red[256];
    int b = blockIdx.x;
    float s = 0.0f;
    for (int h = threadIdx.x; h < Hd; h += 256)
        s += ctx[b * Hd + h] * ws[h] + ms[b * Hd + h] * ws[Hd + h];
    red[threadIdx.x] = s; __syncthreads();
    for (int k = 128; k > 0; k >>= 1) {
        if (threadIdx.x < k) red[threadIdx.x] += red[threadIdx.x + k];
        __syncthreads();
    }
    if (threadIdx.x == 0) {
        float l = red[0] + bs[0];
        out[STOPL_OFF + b * NT + t] = l;
        out[STOPP_OFF + b * NT + t] = 1.0f / (1.0f + expf(-l));
    }
}

__global__ void gru_kernel(const float* __restrict__ gi, const float* __restrict__ gh,
                           const float* __restrict__ bih, const float* __restrict__ bhh,
                           float* __restrict__ state) {
    int idx = blockIdx.x * blockDim.x + threadIdx.x;
    if (idx >= Bn * Hd) return;
    int b = idx / Hd, h = idx % Hd;
    const float* gib = gi + (long)b * 3 * Hd;
    const float* ghb = gh + (long)b * 3 * Hd;
    float ir = gib[h] + bih[h],               hr = ghb[h] + bhh[h];
    float iz = gib[Hd + h] + bih[Hd + h],     hz = ghb[Hd + h] + bhh[Hd + h];
    float in_ = gib[2 * Hd + h] + bih[2 * Hd + h];
    float hn  = ghb[2 * Hd + h] + bhh[2 * Hd + h];
    float r = 1.0f / (1.0f + expf(-(ir + hr)));
    float z = 1.0f / (1.0f + expf(-(iz + hz)));
    float n = tanhf(in_ + r * hn);
    state[idx] = (1.0f - z) * n + z * state[idx];
}

__global__ void chain_kernel(float* __restrict__ out) {
    int b = threadIdx.x;
    if (b >= Bn) return;
    int len = NT; bool hit = false;
    for (int t = 0; t < NT; ++t) {
        if (!hit && out[STOPP_OFF + b * NT + t] >= 0.5f) { len = t + 1; hit = true; }
    }
    out[CL_OFF + b] = (float)len;
}

// ------------------------------- orchestration ------------------------------

extern "C" void kernel_launch(void* const* d_in, const int* in_sizes, int n_in,
                              void* d_out, int out_size, void* d_ws, size_t ws_size,
                              hipStream_t stream) {
    const float* logic  = (const float*)d_in[0];
    const float* prompt = (const float*)d_in[1];
    const float* cbk    = (const float*)d_in[2];
    const float* w_init = (const float*)d_in[3];
    const float* w_q    = (const float*)d_in[4];
    const float* w_k    = (const float*)d_in[5];
    const float* w_v    = (const float*)d_in[6];
    const float* sq     = (const float*)d_in[7];
    const float* w_sq   = (const float*)d_in[8];
    const float* w_op   = (const float*)d_in[9];
    const float* w_gate = (const float*)d_in[10];
    const float* b_gate = (const float*)d_in[11];
    const float* w_stop = (const float*)d_in[12];
    const float* b_stop = (const float*)d_in[13];
    const float* wih    = (const float*)d_in[14];
    const float* whh    = (const float*)d_in[15];
    const float* bih    = (const float*)d_in[16];
    const float* bhh    = (const float*)d_in[17];
    float* out = (float*)d_out;

    float* w = (float*)d_ws;
    size_t off = 0;
    auto alloc = [&](size_t n) { float* p = w + off; off += n; return p; };

    float* pk      = alloc((size_t)Bn * LP * Hd);      // 8*512*768
    float* pv      = alloc((size_t)Bn * LP * Hd);
    float* ps      = alloc(Bn * Hd);
    float* ls      = alloc(Bn * Hd);
    float* cc      = alloc(Bn * 2 * Hd);
    float* state   = alloc(Bn * Hd);
    float* state_q = alloc(Bn * Hd);
    float* ctx_sc  = alloc(Bn * LP);
    float* ctx     = alloc(Bn * Hd);
    float* op_pre  = alloc(Bn * Hd);
    float* opn     = alloc(Bn);
    float* cbn     = alloc(CN);
    float* seed    = alloc(Bn * NS * Hd);
    float* slot_q  = alloc(Bn * NS * Hd);
    float* attn    = alloc(Bn * NS * LP);
    float* tens    = alloc(Bn * NS * Hd);
    float* probs   = alloc(Bn * NS);
    float* mask    = alloc(Bn * NS);
    float* cnt     = alloc(Bn);
    float* ms      = alloc(Bn * Hd);
    float* gi      = alloc(Bn * 3 * Hd);
    float* gh      = alloc(Bn * 3 * Hd);
    int*   opidx   = (int*)alloc(Bn);

    const float invsH = 1.0f / sqrtf((float)Hd);

    // --- init phase ---
    mean_kernel<<<(Bn * Hd + 255) / 256, 256, 0, stream>>>(prompt, ps, LP);
    mean_kernel<<<(Bn * Hd + 255) / 256, 256, 0, stream>>>(logic, ls, LL);
    concat_kernel<<<(Bn * 2 * Hd + 255) / 256, 256, 0, stream>>>(ps, ls, cc);
    // state = tanh(cc @ w_init)  (8 x 1536) * (1536 x 768)
    gemm<false, 1>(stream, cc, w_init, state, Bn, Hd, 2 * Hd, 2 * Hd, Hd, Hd, 0, 0, 0, 1);
    // pk / pv : big GEMMs (4096 x 768) * (768 x 768) on the WMMA pipe
    gemm<false, 0>(stream, prompt, w_k, pk, Bn * LP, Hd, Hd, Hd, Hd, Hd, 0, 0, 0, 1);
    gemm<false, 0>(stream, prompt, w_v, pv, Bn * LP, Hd, Hd, Hd, Hd, Hd, 0, 0, 0, 1);
    // -||cb_n||^2 once
    neg_sumsq_kernel<<<CN, 256, 0, stream>>>(cbk, cbn, Hd);

    // --- 4 serial chain steps ---
    for (int t = 0; t < NT; ++t) {
        // state_q = state @ w_q
        gemm<false, 0>(stream, state, w_q, state_q, Bn, Hd, Hd, Hd, Hd, Hd, 0, 0, 0, 1);
        // ctx scores: batched (1 x 768) * pk[b]^T (768 x 512), scaled 1/sqrt(H)
        gemm<true, 0>(stream, state_q, pk, ctx_sc, 1, LP, Hd, Hd, Hd, LP,
                      Hd, (long)LP * Hd, LP, Bn, invsH);
        softmax_kernel<<<Bn, 128, 0, stream>>>(ctx_sc, LP);
        // state_ctx = probs @ pv[b]
        gemm<false, 0>(stream, ctx_sc, pv, ctx, 1, Hd, LP, LP, Hd, Hd,
                       LP, (long)LP * Hd, Hd, Bn);
        // op_pre = ctx @ w_op_pre
        gemm<false, 0>(stream, ctx, w_op, op_pre, Bn, Hd, Hd, Hd, Hd, Hd, 0, 0, 0, 1);
        neg_sumsq_kernel<<<Bn, 256, 0, stream>>>(op_pre, opn, Hd);
        // op_logits = 2*op_pre@cb^T - ||op_pre||^2 - ||cb||^2, straight into d_out
        gemm<true, 0>(stream, op_pre, cbk, out + OPL_OFF + t * CN,
                      Bn, CN, Hd, Hd, Hd, NT * CN, 0, 0, 0, 1, 2.0f, opn, cbn);
        argmax_kernel<<<Bn, 256, 0, stream>>>(out + OPL_OFF + t * CN, opidx, CN, NT * CN);
        // slots
        slot_seed_kernel<<<(Bn * NS * Hd + 255) / 256, 256, 0, stream>>>(ctx, sq, seed);
        gemm<false, 0>(stream, seed, w_sq, slot_q, Bn * NS, Hd, Hd, Hd, Hd, Hd, 0, 0, 0, 1);
        gemm<true, 0>(stream, slot_q, pk, attn, NS, LP, Hd, Hd, Hd, LP,
                      (long)NS * Hd, (long)LP * Hd, (long)NS * LP, Bn, invsH);
        softmax_kernel<<<Bn * NS, 128, 0, stream>>>(attn, LP);
        gemm<false, 0>(stream, attn, pv, tens, NS, Hd, LP, LP, Hd, Hd,
                       (long)NS * LP, (long)LP * Hd, (long)NS * Hd, Bn);
        slot_gate_kernel<<<Bn * NS, 256, 0, stream>>>(seed, w_gate, b_gate, probs);
        slot_mask_kernel<<<1, 32, 0, stream>>>(probs, mask, cnt);
        matrix_summary_kernel<<<(Bn * Hd + 255) / 256, 256, 0, stream>>>(
            cbk, opidx, tens, mask, cnt, ms, out, t);
        stop_kernel<<<Bn, 256, 0, stream>>>(ctx, ms, w_stop, b_stop, out, t);
        // GRU: gi = ms @ wih^T, gh = state @ whh^T
        gemm<true, 0>(stream, ms, wih, gi, Bn, 3 * Hd, Hd, Hd, Hd, 3 * Hd, 0, 0, 0, 1);
        gemm<true, 0>(stream, state, whh, gh, Bn, 3 * Hd, Hd, Hd, Hd, 3 * Hd, 0, 0, 0, 1);
        gru_kernel<<<(Bn * Hd + 255) / 256, 256, 0, stream>>>(gi, gh, bih, bhh, state);
    }
    chain_kernel<<<1, 32, 0, stream>>>(out);
}
